// RNN_22325240004797
// MI455X (gfx1250) — matrix operand order repaired and checked by reading, not compiled, not run on hardware
//
#include <hip/hip_runtime.h>
#include <cstdint>
#include <cstddef>

// Problem sizes (fixed by the reference)
#define B_   32
#define S_   1024
#define IN_  32
#define H_   512
#define OUT_ 32
#define NWG  16   // persistent workgroups for the recurrence (H_/32)

// Padded LDS pitches (avoid 16-way bank conflicts: 520 halfs = 1040 B,
// 1040/4 mod 64 = 4 -> consecutive rows/cols start 4 banks apart).
#define WPITCH 520

typedef __attribute__((ext_vector_type(16))) _Float16 v16h;
typedef __attribute__((ext_vector_type(8)))  float    v8f;

struct alignas(16) U128 { unsigned int v[4]; };
union Frag { v16h h; U128 q[2]; };

// ---------------------------------------------------------------------------
// CDNA5 async global->LDS copy (ASYNCcnt-tracked, no VGPR round trip).
// VDST VGPR carries the LDS byte address; flat pointers to LDS carry the LDS
// offset in their low 32 bits (ISA 10.2).
// ---------------------------------------------------------------------------
__device__ __forceinline__ void async_copy_b128(void* lds_dst, const void* gsrc) {
  unsigned loff = (unsigned)(uintptr_t)lds_dst;
  asm volatile("global_load_async_to_lds_b128 %0, %1, off"
               :: "v"(loff), "v"(gsrc) : "memory");
}
__device__ __forceinline__ void wait_async() {
  asm volatile("s_wait_asynccnt 0x0" ::: "memory");
}

// ---------------------------------------------------------------------------
// Fragment loaders for V_WMMA_F32_16X16X32_F16 (wave32).
// A (16x32 f16): lane L<16 holds row L, K-chunks {0..7} and {16..23};
//                lane L+16 holds row L, K-chunks {8..15} and {24..31}.
// B (32x16 f16): lane n<16 holds col n, K-chunks {0..7} and {16..23};
//                lane n+16 holds col n, K-chunks {8..15} and {24..31}.
// ---------------------------------------------------------------------------
__device__ __forceinline__ v16h load_a_frag(const _Float16* base, size_t rstride,
                                            int mt, int kt, int lane) {
  const int lm = lane & 15;
  const _Float16* p = base + (size_t)(mt * 16 + lm) * rstride + kt * 32 + ((lane < 16) ? 0 : 8);
  Frag f;
  f.q[0] = *(const U128*)(p);
  f.q[1] = *(const U128*)(p + 16);
  return f.h;
}

__device__ __forceinline__ v16h load_b_frag(const _Float16* base, size_t cstride,
                                            int colbase, int kt, int lane) {
  const int lm = lane & 15;
  const _Float16* p = base + (size_t)(colbase + lm) * cstride + kt * 32 + ((lane < 16) ? 0 : 8);
  Frag f;
  f.q[0] = *(const U128*)(p);
  f.q[1] = *(const U128*)(p + 16);
  return f.h;
}

// ---------------------------------------------------------------------------
// Persistent GRU layer kernel.
// Grid: NWG workgroups x 256 threads (8 wave32).  Workgroup w owns hidden
// units [w*32, w*32+32).  Its Wh slice (512 x 96 f16, column-major, padded)
// lives in LDS; h_t and x_t are staged into LDS each step via async copies;
// Wi B-fragments stream from global (L2-hot).  Per step, 16 accumulator
// tiles (q in {r, z, xn, hn}) x (2 M-tiles x 2 N-tiles); waves 0-3 compute
// (r, xn), waves 4-7 compute (z, hn).  Gate math in f32, then h_new is
// exchanged through ping-pong global f16 buffers with a per-step
// atomic-counter grid barrier.
// ---------------------------------------------------------------------------
__global__ void __launch_bounds__(256)
gru_persistent(const _Float16* __restrict__ x, int kin,
               const _Float16* __restrict__ WiT,   // (3H, kin) col-major of Wi
               const _Float16* __restrict__ WhT,   // (3H, H)   col-major of Wh
               const float* __restrict__ bi,       // (3H) f32
               const float* __restrict__ bhn,      // (H)  f32
               _Float16* __restrict__ hb,          // 2 * B_*H_ halfs, ping-pong
               unsigned int* __restrict__ cnt,     // S_ step counters (zeroed)
               _Float16* __restrict__ ys)          // (B,S,H) f16 output
{
  extern __shared__ char smem[];
  _Float16* whs = (_Float16*)smem;                         // [96][WPITCH] = 99840 B
  float*    cts = (float*)(smem + 99840);                  // [16][16][16] = 16384 B
  _Float16* hs  = (_Float16*)(smem + 99840 + 16384);       // [32][WPITCH] = 33280 B
  _Float16* xs  = (_Float16*)(smem + 99840 + 16384 + 33280); // [32][kin+8]

  const int tid   = threadIdx.x;
  const int wg    = blockIdx.x;
  const int lane  = tid & 31;
  const int wave  = tid >> 5;
  const int mt    = (wave >> 1) & 1;
  const int nt    = wave & 1;
  const int group = wave >> 2;

  // Stage this WG's Wh columns into LDS via async copies (one-time).
  for (int i = tid; i < 96 * 64; i += 256) {
    const int cl = i >> 6, kv = i & 63;
    const int q3 = cl >> 5, j = cl & 31;
    const _Float16* g = WhT + (size_t)(q3 * 512 + wg * 32 + j) * H_ + kv * 8;
    async_copy_b128(whs + cl * WPITCH + kv * 8, g);
  }
  wait_async();
  __syncthreads();

  const int    nkt_in   = kin >> 5;
  const int    xpitch   = kin + 8;
  const int    xch      = kin >> 3;        // 16B chunks per x row
  const size_t xrstride = (size_t)S_ * (size_t)kin;

  for (int t = 0; t < S_; ++t) {
    const _Float16* hcur = hb + (size_t)(t & 1) * (B_ * H_);
    _Float16*       hnxt = hb + (size_t)((t + 1) & 1) * (B_ * H_);
    const _Float16* xt   = x + (size_t)t * (size_t)kin;

    // Async-stage h_t (32x512) and x_t (32xkin) into LDS.
    for (int i = tid; i < 32 * 64; i += 256) {
      const int b = i >> 6, c8 = i & 63;
      async_copy_b128(hs + b * WPITCH + c8 * 8, hcur + b * H_ + c8 * 8);
    }
    for (int i = tid; i < 32 * xch; i += 256) {
      const int b = i / xch, c8 = i % xch;
      async_copy_b128(xs + b * xpitch + c8 * 8, xt + (size_t)b * xrstride + c8 * 8);
    }
    wait_async();
    __syncthreads();

    // Two jobs per wave: q=0(r)/2(xn) for group 0, q=1(z)/3(hn) for group 1.
    for (int ji = 0; ji < 2; ++ji) {
      const int q    = group ? (ji ? 3 : 1) : (ji ? 2 : 0);
      const int gate = (q < 2) ? q : 2;
      v8f acc = {};
      if (q != 3) {  // x @ Wi contribution (fused into r/z accumulators)
        const int colbase = gate * 512 + wg * 32 + nt * 16;
        for (int kt = 0; kt < nkt_in; ++kt) {
          v16h a = load_a_frag(xs, (size_t)xpitch, mt, kt, lane);
          v16h b = load_b_frag(WiT, (size_t)kin, colbase, kt, lane);
          acc = __builtin_amdgcn_wmma_f32_16x16x32_f16(false, a, false, b,
                                                       (short)0, acc, false, false);
        }
      }
      if (q != 2) {  // h @ Wh contribution (A and B fragments from LDS)
        const int colbase = gate * 32 + nt * 16;
        for (int kt = 0; kt < 16; ++kt) {
          v16h a = load_a_frag(hs, (size_t)WPITCH, mt, kt, lane);
          v16h b = load_b_frag(whs, (size_t)WPITCH, colbase, kt, lane);
          acc = __builtin_amdgcn_wmma_f32_16x16x32_f16(false, a, false, b,
                                                       (short)0, acc, false, false);
        }
      }
      // Stash C tile to LDS for the cross-wave gate epilogue.
      float* ct = cts + ((q * 2 + mt) * 2 + nt) * 256;
      const int lm = lane & 15;
      const int ro = (lane < 16) ? 0 : 8;
      #pragma unroll
      for (int r = 0; r < 8; ++r) ct[(ro + r) * 16 + lm] = acc[r];
    }
    __syncthreads();

    // Gate epilogue: 32 batch x 32 owned hidden units = 1024 elems / 256 thr.
    #pragma unroll
    for (int i = 0; i < 4; ++i) {
      const int e  = i * 256 + tid;
      const int b  = e >> 5, jl = e & 31;
      const int bm = b >> 4, jn = jl >> 4;
      const int off = (b & 15) * 16 + (jl & 15);
      const float Cr  = cts[((0 * 2 + bm) * 2 + jn) * 256 + off];
      const float Cz  = cts[((1 * 2 + bm) * 2 + jn) * 256 + off];
      const float Cxn = cts[((2 * 2 + bm) * 2 + jn) * 256 + off];
      const float Chn = cts[((3 * 2 + bm) * 2 + jn) * 256 + off];
      const int jg = wg * 32 + jl;
      const float r  = 1.f / (1.f + __expf(-(Cr + bi[jg])));
      const float z  = 1.f / (1.f + __expf(-(Cz + bi[512 + jg])));
      const float n  = tanhf(Cxn + bi[1024 + jg] + r * (Chn + bhn[jg]));
      const float hp = (float)hs[b * WPITCH + jg];
      const float hnew = (1.f - z) * n + z * hp;
      hnxt[b * H_ + jg] = (_Float16)hnew;
      ys[((size_t)b * S_ + t) * H_ + jg] = (_Float16)hnew;
    }

    // Grid-wide barrier: monotonic per-step counter (no reset races).
    __threadfence();
    if (tid == 0) {
      __hip_atomic_fetch_add(&cnt[t], 1u, __ATOMIC_RELEASE, __HIP_MEMORY_SCOPE_AGENT);
      while (__hip_atomic_load(&cnt[t], __ATOMIC_ACQUIRE, __HIP_MEMORY_SCOPE_AGENT) < NWG) {
        __builtin_amdgcn_s_sleep(4);
      }
    }
    __syncthreads();
    __threadfence();
  }
}

// ---------------------------------------------------------------------------
// Head GEMM: out(32768x32) = ys1(32768x512) @ Wl(512x32) + bl, f32 out.
// One wave per 16-row M-tile, two N-tiles per wave.
// ---------------------------------------------------------------------------
__global__ void __launch_bounds__(128)
head_kernel(const _Float16* __restrict__ ys1,
            const _Float16* __restrict__ WlT,   // (32, 512) col-major of Wl
            const float* __restrict__ bl,
            float* __restrict__ out)
{
  const int lane = threadIdx.x & 31;
  const int wave = threadIdx.x >> 5;
  const int mt   = blockIdx.x * 4 + wave;  // 0..2047
  v8f a0 = {}, a1 = {};
  for (int kt = 0; kt < 16; ++kt) {
    v16h a  = load_a_frag(ys1, (size_t)H_, mt, kt, lane);
    v16h b0 = load_b_frag(WlT, (size_t)H_, 0, kt, lane);
    v16h b1 = load_b_frag(WlT, (size_t)H_, 16, kt, lane);
    a0 = __builtin_amdgcn_wmma_f32_16x16x32_f16(false, a, false, b0, (short)0, a0, false, false);
    a1 = __builtin_amdgcn_wmma_f32_16x16x32_f16(false, a, false, b1, (short)0, a1, false, false);
  }
  const int lm = lane & 15;
  const int ro = (lane < 16) ? 0 : 8;
  const float b0 = bl[lm], b1 = bl[16 + lm];
  #pragma unroll
  for (int r = 0; r < 8; ++r) {
    const size_t row = (size_t)mt * 16 + ro + r;
    out[row * OUT_ + lm]      = a0[r] + b0;
    out[row * OUT_ + 16 + lm] = a1[r] + b1;
  }
}

// ---------------------------------------------------------------------------
// Utility kernels
// ---------------------------------------------------------------------------
__global__ void cvt_f16_kernel(const float* __restrict__ src, _Float16* __restrict__ dst, int n) {
  int i = blockIdx.x * blockDim.x + threadIdx.x;
  if (i < n) dst[i] = (_Float16)src[i];
}

// dst (N, K) col-major f16  <-  src (K, N) row-major f32
__global__ void transpose_f16_kernel(const float* __restrict__ src, _Float16* __restrict__ dst,
                                     int K, int N) {
  int i = blockIdx.x * blockDim.x + threadIdx.x;
  if (i < K * N) {
    int k = i / N, n = i % N;
    dst[(size_t)n * K + k] = (_Float16)src[i];
  }
}

__global__ void init_layer_kernel(unsigned int* __restrict__ hbuf32, unsigned int* __restrict__ cnt) {
  int i = blockIdx.x * blockDim.x + threadIdx.x;
  if (i < 16384) hbuf32[i] = 0u;  // 2 * 32 * 512 halfs
  if (i < 1024)  cnt[i]    = 0u;  // S_ step counters
}

// ---------------------------------------------------------------------------
// Launch
// ---------------------------------------------------------------------------
extern "C" void kernel_launch(void* const* d_in, const int* in_sizes, int n_in,
                              void* d_out, int out_size, void* d_ws, size_t ws_size,
                              hipStream_t stream) {
  (void)in_sizes; (void)n_in; (void)out_size; (void)ws_size;
  const float* u    = (const float*)d_in[0];
  const float* Wi0  = (const float*)d_in[1];
  const float* bi0  = (const float*)d_in[2];
  const float* Wh0  = (const float*)d_in[3];
  const float* bhn0 = (const float*)d_in[4];
  const float* Wi1  = (const float*)d_in[5];
  const float* bi1  = (const float*)d_in[6];
  const float* Wh1  = (const float*)d_in[7];
  const float* bhn1 = (const float*)d_in[8];
  const float* Wl   = (const float*)d_in[9];
  const float* bl   = (const float*)d_in[10];
  float* outp = (float*)d_out;

  // Workspace layout (all 256B-aligned offsets)
  char* ws = (char*)d_ws;
  _Float16* u16  = (_Float16*)(ws + 0);          //  2 MB : (B,S,IN) f16
  _Float16* WiT0 = (_Float16*)(ws + 2097152);    // 96 KB : (1536,32)
  _Float16* WhT0 = (_Float16*)(ws + 2195456);    // 1.5MB : (1536,512)
  _Float16* WiT1 = (_Float16*)(ws + 3768320);    // 1.5MB : (1536,512)
  _Float16* WhT1 = (_Float16*)(ws + 5341184);    // 1.5MB : (1536,512)
  _Float16* WlT  = (_Float16*)(ws + 6914048);    // 32 KB : (32,512)
  _Float16* hbuf = (_Float16*)(ws + 6946816);    // 64 KB : 2x(32,512) ping-pong
  unsigned int* cnt = (unsigned int*)(ws + 7012352); // 8 KB : 2x1024 counters
  _Float16* ys0  = (_Float16*)(ws + 7020544);    // 32 MB : (B,S,H) f16
  _Float16* ys1  = (_Float16*)(ws + 40574976);   // 32 MB : (B,S,H) f16

  // Precision/layout prep: f16 conversions + column-major weight transposes
  cvt_f16_kernel<<<dim3((B_*S_*IN_ + 255)/256), dim3(256), 0, stream>>>(u, u16, B_*S_*IN_);
  transpose_f16_kernel<<<dim3((IN_*3*H_ + 255)/256), dim3(256), 0, stream>>>(Wi0, WiT0, IN_, 3*H_);
  transpose_f16_kernel<<<dim3((H_*3*H_ + 255)/256),  dim3(256), 0, stream>>>(Wh0, WhT0, H_, 3*H_);
  transpose_f16_kernel<<<dim3((H_*3*H_ + 255)/256),  dim3(256), 0, stream>>>(Wi1, WiT1, H_, 3*H_);
  transpose_f16_kernel<<<dim3((H_*3*H_ + 255)/256),  dim3(256), 0, stream>>>(Wh1, WhT1, H_, 3*H_);
  transpose_f16_kernel<<<dim3((H_*OUT_ + 255)/256),  dim3(256), 0, stream>>>(Wl, WlT, H_, OUT_);

  // Dynamic LDS: Wh slice (padded) + C tiles + h_t stage + x_t stage
  const size_t smem_base = 99840 + 16384 + 33280;     // whs + cts + hs
  const size_t smem0 = smem_base + (size_t)64 * (IN_ + 8);   // + xs (layer 0)
  const size_t smem1 = smem_base + (size_t)64 * (H_  + 8);   // + xs (layer 1)

  // Layer 0 (kin = 32, input projection fused into the recurrence)
  init_layer_kernel<<<dim3(64), dim3(256), 0, stream>>>((unsigned int*)hbuf, cnt);
  gru_persistent<<<dim3(NWG), dim3(256), smem0, stream>>>(u16, IN_, WiT0, WhT0,
                                                          bi0, bhn0, hbuf, cnt, ys0);
  // Layer 1 (kin = 512)
  init_layer_kernel<<<dim3(64), dim3(256), 0, stream>>>((unsigned int*)hbuf, cnt + 1024);
  gru_persistent<<<dim3(NWG), dim3(256), smem1, stream>>>(ys0, H_, WiT1, WhT1,
                                                          bi1, bhn1, hbuf, cnt + 1024, ys1);
  // Linear head
  head_kernel<<<dim3(512), dim3(128), 0, stream>>>(ys1, WlT, bl, outp);
}
